// EngramAttention_14886356648013
// MI455X (gfx1250) — matrix operand order repaired and checked by reading, not compiled, not run on hardware
//
#include <hip/hip_runtime.h>
#include <hip/hip_bf16.h>

typedef _Float16 h16;
typedef __attribute__((ext_vector_type(8)))  _Float16 v8h;
typedef __attribute__((ext_vector_type(16))) _Float16 v16h;
typedef __attribute__((ext_vector_type(8)))  float    v8f;
typedef __attribute__((ext_vector_type(4)))  unsigned int u32x4;
typedef __attribute__((ext_vector_type(4)))  int i32x4;
typedef __attribute__((ext_vector_type(8)))  int i32x8;

// ---------------------------------------------------------------------------
// WMMA fragment helpers (wave32, 16x16x32 f16 -> f32)
//
// A (16x32, MxK): lane L holds row M=L&15.
//   elems 0..7  = K: (L>=16?8:0)+0..7      elems 8..15 = K: (L>=16?8:0)+16..23
// B (32x16, KxN): lane L holds col N=L&15.
//   elems 0..7  = K: (L>=16?8:0)+0..7      elems 8..15 = K: (L>=16?8:0)+16..23   (LDS tile stored K-major: [N][K])
// C/D (16x16 f32): lane L elem r -> row r+((L>=16)?8:0), col L&15.
// ---------------------------------------------------------------------------
__device__ inline v16h frag_a_lds(const h16* base, int ld) {
    const int lane = threadIdx.x & 31;
    const int m  = lane & 15;
    const int kh = (lane >> 4) << 3;          // 0 or 8
    const h16* p = base + m * ld + kh;
    v8h lo = *(const v8h*)(p);                // K = kh .. kh+7
    v8h hi = *(const v8h*)(p + 16);           // K = kh+16 .. kh+23
    v16h r;
#pragma unroll
    for (int i = 0; i < 8; ++i) { r[i] = lo[i]; r[i + 8] = hi[i]; }
    return r;
}

__device__ inline v16h frag_b_lds(const h16* base, int ld) {
    const int lane = threadIdx.x & 31;
    const int n  = lane & 15;
    const int kb = (lane >> 4) << 3;          // 0 or 8
    const h16* p = base + n * ld + kb;
    v8h lo = *(const v8h*)(p);
    v8h hi = *(const v8h*)(p + 16);
    v16h r;
#pragma unroll
    for (int i = 0; i < 8; ++i) { r[i] = lo[i]; r[i + 8] = hi[i]; }
    return r;
}

__device__ inline v8f wmma_f16(v16h a, v16h b, v8f c) {
    return __builtin_amdgcn_wmma_f32_16x16x32_f16(
        /*neg_a=*/false, a, /*neg_b=*/false, b,
        /*c_mod=*/(short)0, c, /*reuse_a=*/false, /*reuse_b=*/false);
}

// ---------------------------------------------------------------------------
// Tensor Data Mover: async 2D f16 tile load Global -> LDS with row padding.
// D# per CDNA5 ISA:
//   group0: count=1 | lds_addr[63:32] | global_addr[120:64] | type=2[127:126]
//   group1: data_size=1(2B)[17:16], pad_enable[20], pad_interval[24:22],
//           pad_amount[31:25], tensor_dim0[79:48], tensor_dim1[111:80],
//           tile_dim0[127:112], tile_dim1[143:128], tensor_dim0_stride[207:160]
// Issued by one wave; tracked with TENSORcnt.
// ---------------------------------------------------------------------------
__device__ inline void tdm_load_2d(unsigned lds_byte_off, const void* gptr,
                                   unsigned tdim0, unsigned tdim1,
                                   unsigned tile0, unsigned tile1,
                                   unsigned long long stride0,
                                   unsigned pad_iv, unsigned pad_amt) {
    unsigned long long ga = (unsigned long long)(size_t)gptr;
    u32x4 g0;
    g0[0] = 1u;                                               // count=1 (valid)
    g0[1] = lds_byte_off;                                     // lds_addr
    g0[2] = (unsigned)(ga & 0xffffffffull);                   // global_addr lo
    g0[3] = (unsigned)((ga >> 32) & 0x01ffffffull) | (2u << 30);  // hi | type=2
    i32x8 g1;
    g1[0] = (int)((1u << 16) | (1u << 20) | (pad_iv << 22) | (pad_amt << 25));
    g1[1] = (int)((tdim0 & 0xffffu) << 16);
    g1[2] = (int)(((tdim0 >> 16) & 0xffffu) | ((tdim1 & 0xffffu) << 16));
    g1[3] = (int)(((tdim1 >> 16) & 0xffffu) | ((tile0 & 0xffffu) << 16));
    g1[4] = (int)(tile1 & 0xffffu);                           // tile_dim2 = 0
    g1[5] = (int)(unsigned)(stride0 & 0xffffffffull);
    g1[6] = (int)(unsigned)((stride0 >> 32) & 0xffffull);
    g1[7] = 0;
    i32x4 gz = {0, 0, 0, 0};
#if defined(__clang_major__) && (__clang_major__ >= 23)
    i32x8 gz8 = {};
    __builtin_amdgcn_tensor_load_to_lds(g0, g1, gz, gz, gz8, 0);
#else
    __builtin_amdgcn_tensor_load_to_lds(g0, g1, gz, gz, 0);
#endif
}

// ---------------------------------------------------------------------------
// Kernel 1: qkv = x @ Wqkv + bqkv  (M=4096, K=1024, N=3072)
// f32 inputs need conversion, so staging stays manual (VALU cvt).
// Block tile 128x64, 8 waves (4x2), each wave 32x32 (2x2 WMMA tiles).
// ---------------------------------------------------------------------------
__global__ __launch_bounds__(256) void qkv_gemm_kernel(
    const float* __restrict__ x, const float* __restrict__ Wqkv,
    const float* __restrict__ bqkv,
    h16* __restrict__ qbuf, h16* __restrict__ kbuf, h16* __restrict__ vbuf)
{
    __shared__ h16 smem[128 * 40 + 64 * 40];
    h16* As = smem;                 // [m][k], ld=40 halves
    h16* Bs = smem + 128 * 40;      // [n][k] transposed, ld=40

    const int tid  = threadIdx.x;
    const int wid  = tid >> 5, lane = tid & 31;
    const int m0   = blockIdx.y * 128;
    const int n0   = blockIdx.x * 64;
    const int wm   = (wid & 3) * 32;
    const int wn   = (wid >> 2) * 32;

    v8f acc[2][2] = {};

    for (int k0 = 0; k0 < 1024; k0 += 32) {
        // stage A: 128x32 f32 -> f16   (1024 float4, 4 per thread)
#pragma unroll
        for (int i = 0; i < 4; ++i) {
            int idx = tid + i * 256;
            int row = idx >> 3, c4 = (idx & 7) * 4;
            float4 v = *(const float4*)(x + (size_t)(m0 + row) * 1024 + k0 + c4);
            h16* d = As + row * 40 + c4;
            d[0] = (h16)v.x; d[1] = (h16)v.y; d[2] = (h16)v.z; d[3] = (h16)v.w;
        }
        // stage B transposed: Wqkv[k0..+31][n0..+63] -> Bs[n][k]
#pragma unroll
        for (int i = 0; i < 2; ++i) {
            int idx = tid + i * 256;
            int kr = idx >> 4, nf = (idx & 15) * 4;
            float4 v = *(const float4*)(Wqkv + (size_t)(k0 + kr) * 3072 + n0 + nf);
            Bs[(nf + 0) * 40 + kr] = (h16)v.x;
            Bs[(nf + 1) * 40 + kr] = (h16)v.y;
            Bs[(nf + 2) * 40 + kr] = (h16)v.z;
            Bs[(nf + 3) * 40 + kr] = (h16)v.w;
        }
        if (k0 + 32 < 1024)
            __builtin_prefetch(x + (size_t)(m0 + (tid >> 3)) * 1024 + k0 + 32, 0, 0);
        __syncthreads();

        v16h a0 = frag_a_lds(As + (wm +  0) * 40, 40);
        v16h a1 = frag_a_lds(As + (wm + 16) * 40, 40);
        v16h b0 = frag_b_lds(Bs + (wn +  0) * 40, 40);
        v16h b1 = frag_b_lds(Bs + (wn + 16) * 40, 40);
        acc[0][0] = wmma_f16(a0, b0, acc[0][0]);
        acc[0][1] = wmma_f16(a0, b1, acc[0][1]);
        acc[1][0] = wmma_f16(a1, b0, acc[1][0]);
        acc[1][1] = wmma_f16(a1, b1, acc[1][1]);
        __syncthreads();
    }

    // epilogue: bias + scatter to [B,H,T,64] f16 buffers
#pragma unroll
    for (int mi = 0; mi < 2; ++mi)
#pragma unroll
    for (int ni = 0; ni < 2; ++ni) {
#pragma unroll
        for (int r = 0; r < 8; ++r) {
            int gm = m0 + wm + mi * 16 + r + ((lane >> 4) << 3);
            int gn = n0 + wn + ni * 16 + (lane & 15);
            float v = acc[mi][ni][r] + bqkv[gn];
            int s = gn >> 10;              // 0=q 1=k 2=v
            int d = gn & 1023;
            int h = d >> 6, hd = d & 63;
            int b = gm >> 11, t = gm & 2047;
            size_t off = ((size_t)((b * 16 + h) * 2048 + t) << 6) + hd;
            h16* dst = (s == 0) ? qbuf : (s == 1 ? kbuf : vbuf);
            dst[off] = (h16)v;
        }
    }
}

// ---------------------------------------------------------------------------
// Kernel 2: causal flash attention per (b,h). Q block 128 rows, 8 waves own
// 16 rows each. Key tiles of 64; QK^T and PV on WMMA, online softmax.
// Q and K tiles staged via TDM (tensor_load_to_lds) with native row padding.
// ---------------------------------------------------------------------------
__global__ __launch_bounds__(256) void attn_kernel(
    const h16* __restrict__ qbuf, const h16* __restrict__ kbuf,
    const h16* __restrict__ vbuf, h16* __restrict__ abuf)
{
    // single shared block so LDS byte offsets for the TDM are known (base 0)
    __shared__ h16 smem[128 * 72 + 64 * 72 + 64 * 72 + 8 * 16 * 72];
    h16* Qs = smem;                                 // [qrow][hd], ld=72
    h16* Ks = smem + 128 * 72;                      // [key][hd],  ld=72
    h16* Vs = smem + 128 * 72 + 64 * 72;            // transposed [hd][key]
    h16* Ps = smem + 128 * 72 + 2 * 64 * 72;        // per-wave P [16][64]
    const unsigned QS_OFF = 0u;
    const unsigned KS_OFF = (unsigned)(128 * 72 * sizeof(h16));

    const int tid = threadIdx.x, wid = tid >> 5, lane = tid & 31;
    const int bh  = blockIdx.y;                 // b*16 + h
    const int q0  = blockIdx.x * 128;
    const size_t base = (size_t)bh * 2048 * 64;

    // stage Q via TDM: 64-wide rows (128B = 32 dwords), pad 4 dwords -> ld 72
    if (wid == 0) {
        tdm_load_2d(QS_OFF, qbuf + base + (size_t)q0 * 64,
                    /*tdim0=*/64, /*tdim1=*/2048, /*tile0=*/64, /*tile1=*/128,
                    /*stride0=*/64, /*pad_iv(32dw)=*/4, /*pad_amt(4dw)=*/3);
        __builtin_amdgcn_s_wait_tensorcnt(0);
    }
    __syncthreads();

    const int wq       = q0 + wid * 16;
    const int wq_end   = wq + 15;
    const int row_base = (lane >> 4) << 3;

    v16h a_q0 = frag_a_lds(Qs + (wid * 16) * 72 +  0, 72);
    v16h a_q1 = frag_a_lds(Qs + (wid * 16) * 72 + 32, 72);

    v8f o[4] = {};
    float m_old[8], l[8];
#pragma unroll
    for (int r = 0; r < 8; ++r) { m_old[r] = -3.0e38f; l[r] = 0.f; }

    const int ktmax = (q0 + 127) >> 6;          // inclusive, causal bound
    for (int kt = 0; kt <= ktmax; ++kt) {
        const int kb = kt * 64;
        // stage K tile via TDM (async; overlaps with manual V staging below)
        if (wid == 0)
            tdm_load_2d(KS_OFF, kbuf + base + (size_t)kb * 64,
                        64, 2048, 64, 64, 64, 4, 3);
        // stage V tile transposed [hd][key] (TDM cannot transpose)
#pragma unroll
        for (int i = 0; i < 2; ++i) {
            int idx = tid + i * 256;
            int key = idx >> 3, c8 = (idx & 7) * 8;
            v8h v = *(const v8h*)(vbuf + base + (size_t)(kb + key) * 64 + c8);
#pragma unroll
            for (int j = 0; j < 8; ++j) Vs[(c8 + j) * 72 + key] = v[j];
        }
        if (wid == 0) __builtin_amdgcn_s_wait_tensorcnt(0);
        __syncthreads();

        if (kb <= wq_end) {
            // S = Q K^T : 4 n-tiles x 2 k-chunks
            v8f s[4] = {};
#pragma unroll
            for (int nt = 0; nt < 4; ++nt) {
                v16h b0 = frag_b_lds(Ks + nt * 16 * 72 +  0, 72);
                v16h b1 = frag_b_lds(Ks + nt * 16 * 72 + 32, 72);
                s[nt] = wmma_f16(a_q0, b0, s[nt]);
                s[nt] = wmma_f16(a_q1, b1, s[nt]);
            }
            // scale (1/sqrt(64)) + causal mask + online softmax
#pragma unroll
            for (int r = 0; r < 8; ++r) {
                const int rq = wq + r + row_base;
                float mx = -3.0e38f;
#pragma unroll
                for (int nt = 0; nt < 4; ++nt) {
                    int kc = kb + nt * 16 + (lane & 15);
                    float sv = s[nt][r] * 0.125f;
                    if (kc > rq) sv = -3.0e38f;
                    s[nt][r] = sv;
                    mx = fmaxf(mx, sv);
                }
                mx = fmaxf(mx, __shfl_xor(mx, 1));
                mx = fmaxf(mx, __shfl_xor(mx, 2));
                mx = fmaxf(mx, __shfl_xor(mx, 4));
                mx = fmaxf(mx, __shfl_xor(mx, 8));
                const float mi    = fmaxf(m_old[r], mx);
                const float alpha = __expf(m_old[r] - mi);
                float rs = 0.f;
#pragma unroll
                for (int nt = 0; nt < 4; ++nt) {
                    float p = __expf(s[nt][r] - mi);
                    s[nt][r] = p;
                    rs += p;
                }
                rs += __shfl_xor(rs, 1);
                rs += __shfl_xor(rs, 2);
                rs += __shfl_xor(rs, 4);
                rs += __shfl_xor(rs, 8);
                l[r]     = l[r] * alpha + rs;
                m_old[r] = mi;
#pragma unroll
                for (int nt = 0; nt < 4; ++nt) o[nt][r] *= alpha;
            }
            // P (C-layout f32) -> LDS -> A-layout f16 frags
            h16* Pw = Ps + wid * 16 * 72;
#pragma unroll
            for (int nt = 0; nt < 4; ++nt)
#pragma unroll
                for (int r = 0; r < 8; ++r)
                    Pw[(r + row_base) * 72 + nt * 16 + (lane & 15)] = (h16)s[nt][r];
            v16h ap0 = frag_a_lds(Pw +  0, 72);
            v16h ap1 = frag_a_lds(Pw + 32, 72);
            // O += P V
#pragma unroll
            for (int nt = 0; nt < 4; ++nt) {
                v16h b0 = frag_b_lds(Vs + nt * 16 * 72 +  0, 72);
                v16h b1 = frag_b_lds(Vs + nt * 16 * 72 + 32, 72);
                o[nt] = wmma_f16(ap0, b0, o[nt]);
                o[nt] = wmma_f16(ap1, b1, o[nt]);
            }
        }
        __syncthreads();
    }

    // finalize: O /= l, write f16 [B,T,D] (D index = h*64 + hd)
    const int b = bh >> 4, h = bh & 15;
#pragma unroll
    for (int r = 0; r < 8; ++r) {
        const float inv = 1.0f / l[r];
        const int trow = wq + r + row_base;
#pragma unroll
        for (int nt = 0; nt < 4; ++nt) {
            int hc = h * 64 + nt * 16 + (lane & 15);
            abuf[(size_t)(b * 2048 + trow) * 1024 + hc] = (h16)(o[nt][r] * inv);
        }
    }
}

// ---------------------------------------------------------------------------
// Kernel 3: out = attn @ Wout + bout  (M=4096, K=1024, N=1024), f32 output.
// A tiles (pure f16 2D copies) staged via TDM; B needs f32->f16, manual.
// ---------------------------------------------------------------------------
__global__ __launch_bounds__(256) void out_gemm_kernel(
    const h16* __restrict__ abuf, const float* __restrict__ Wout,
    const float* __restrict__ bout, float* __restrict__ out)
{
    __shared__ h16 smem[128 * 40 + 64 * 40];
    h16* As = smem;                 // [m][k], ld=40
    h16* Bs = smem + 128 * 40;      // [n][k] transposed, ld=40

    const int tid  = threadIdx.x;
    const int wid  = tid >> 5, lane = tid & 31;
    const int m0   = blockIdx.y * 128;
    const int n0   = blockIdx.x * 64;
    const int wm   = (wid & 3) * 32;
    const int wn   = (wid >> 2) * 32;

    v8f acc[2][2] = {};

    for (int k0 = 0; k0 < 1024; k0 += 32) {
        // stage A via TDM: rows 32 halves (64B = 16 dwords), pad 4 dw -> ld 40
        if (wid == 0)
            tdm_load_2d(0u, abuf + (size_t)m0 * 1024 + k0,
                        /*tdim0=*/32, /*tdim1=*/128, /*tile0=*/32, /*tile1=*/128,
                        /*stride0=*/1024, /*pad_iv(16dw)=*/3, /*pad_amt(4dw)=*/3);
        // stage B transposed: Wout[k0..+31][n0..+63] -> Bs[n][k]
#pragma unroll
        for (int i = 0; i < 2; ++i) {
            int idx = tid + i * 256;
            int kr = idx >> 4, nf = (idx & 15) * 4;
            float4 v = *(const float4*)(Wout + (size_t)(k0 + kr) * 1024 + n0 + nf);
            Bs[(nf + 0) * 40 + kr] = (h16)v.x;
            Bs[(nf + 1) * 40 + kr] = (h16)v.y;
            Bs[(nf + 2) * 40 + kr] = (h16)v.z;
            Bs[(nf + 3) * 40 + kr] = (h16)v.w;
        }
        if (k0 + 32 < 1024)
            __builtin_prefetch(abuf + (size_t)(m0 + (tid >> 2)) * 1024 + k0 + 32, 0, 0);
        if (wid == 0) __builtin_amdgcn_s_wait_tensorcnt(0);
        __syncthreads();

        v16h a0 = frag_a_lds(As + (wm +  0) * 40, 40);
        v16h a1 = frag_a_lds(As + (wm + 16) * 40, 40);
        v16h b0 = frag_b_lds(Bs + (wn +  0) * 40, 40);
        v16h b1 = frag_b_lds(Bs + (wn + 16) * 40, 40);
        acc[0][0] = wmma_f16(a0, b0, acc[0][0]);
        acc[0][1] = wmma_f16(a0, b1, acc[0][1]);
        acc[1][0] = wmma_f16(a1, b0, acc[1][0]);
        acc[1][1] = wmma_f16(a1, b1, acc[1][1]);
        __syncthreads();
    }

#pragma unroll
    for (int mi = 0; mi < 2; ++mi)
#pragma unroll
    for (int ni = 0; ni < 2; ++ni) {
#pragma unroll
        for (int r = 0; r < 8; ++r) {
            int gm = m0 + wm + mi * 16 + r + ((lane >> 4) << 3);
            int gn = n0 + wn + ni * 16 + (lane & 15);
            out[(size_t)gm * 1024 + gn] = acc[mi][ni][r] + bout[gn];
        }
    }
}

// ---------------------------------------------------------------------------
extern "C" void kernel_launch(void* const* d_in, const int* in_sizes, int n_in,
                              void* d_out, int out_size, void* d_ws, size_t ws_size,
                              hipStream_t stream) {
    (void)in_sizes; (void)n_in; (void)out_size; (void)ws_size;
    const float* x    = (const float*)d_in[0];
    const float* Wqkv = (const float*)d_in[1];
    const float* bqkv = (const float*)d_in[2];
    const float* Wout = (const float*)d_in[3];
    const float* bout = (const float*)d_in[4];
    float* out = (float*)d_out;

    const size_t PER_BUF = (size_t)2 * 16 * 2048 * 64;  // B*H*T*hd halves
    h16* qbuf = (h16*)d_ws;
    h16* kbuf = qbuf + PER_BUF;
    h16* vbuf = kbuf + PER_BUF;
    h16* abuf = vbuf + PER_BUF;                         // [B,T,D] f16

    qkv_gemm_kernel<<<dim3(48, 32), 256, 0, stream>>>(x, Wqkv, bqkv, qbuf, kbuf, vbuf);
    attn_kernel   <<<dim3(16, 32), 256, 0, stream>>>(qbuf, kbuf, vbuf, abuf);
    out_gemm_kernel<<<dim3(16, 32), 256, 0, stream>>>(abuf, Wout, bout, out);
}
